// MultiHeadAttention_53145925321274
// MI455X (gfx1250) — matrix operand order repaired and checked by reading, not compiled, not run on hardware
//
#include <hip/hip_runtime.h>

// Problem constants (match reference)
#define D_MODEL   1024
#define NUM_HEADS 16
#define DK        64
#define BATCH     2
#define SEQ       2048
#define M_TOT     (BATCH * SEQ)     // 4096 rows for all projections

typedef __attribute__((ext_vector_type(16))) _Float16     v16h;
typedef __attribute__((ext_vector_type(8)))  float        v8f;
typedef __attribute__((ext_vector_type(4)))  unsigned int u32x4;
typedef __attribute__((ext_vector_type(4)))  int          i32x4;
typedef __attribute__((ext_vector_type(8)))  int          i32x8;

union H16 { v16h v; _Float16 h[16]; u32x4 q[2]; };

__device__ __forceinline__ v8f wmma_f16(v16h a, v16h b, v8f c) {
  // (neg_a, A, neg_b, B, c_mod, C, reuse_a, reuse_b)
  return __builtin_amdgcn_wmma_f32_16x16x32_f16(false, a, false, b, (short)0, c,
                                                false, false);
}

// A-fragment (16x32 f16), row-major source with given stride (halfs).
// Contiguous-k order used here (B matches): lane l holds row l%16; g=l/16: halfs 0..15 = K g*16..g*16+15.
// Two 16B loads per lane.
__device__ __forceinline__ v16h load_a_frag(const _Float16* base, int stride, int lane) {
  int row = lane & 15, g = lane >> 4;
  const _Float16* p = base + (size_t)row * stride + g * 16;
  H16 r;
  r.q[0] = *(const u32x4*)(p);
  r.q[1] = *(const u32x4*)(p + 8);
  return r.v;
}

// B-fragment (32x16 f16) where this lane's 16 K-halfs are contiguous at p.
__device__ __forceinline__ v16h load_b_frag(const _Float16* p) {
  H16 r;
  r.q[0] = *(const u32x4*)(p);
  r.q[1] = *(const u32x4*)(p + 8);
  return r.v;
}

__device__ __forceinline__ void wait_tensorcnt0() {
#if __has_builtin(__builtin_amdgcn_s_wait_tensorcnt)
  __builtin_amdgcn_s_wait_tensorcnt(0);
#else
  asm volatile("s_wait_tensorcnt 0" ::: "memory");
#endif
}

// Issue one TDM 2D tile load: tile_dim0 x tile_dim1 f16 elements, global row
// stride srcStride (halfs), into LDS at ldsOff with 16B padding every 64B
// (row pitch 40 halfs for tile_dim0 == 32).
__device__ __forceinline__ void tdm_load_tile_f16(const _Float16* gsrc,
                                                  unsigned ldsOff,
                                                  unsigned tileW, unsigned tileH,
                                                  unsigned srcStride) {
#if __has_builtin(__builtin_amdgcn_tensor_load_to_lds)
  unsigned long long ga = (unsigned long long)(uintptr_t)gsrc;
  u32x4 g0;
  g0.x = 1u;                                        // count=1, user descriptor
  g0.y = ldsOff;                                    // lds_addr (bytes)
  g0.z = (unsigned)(ga & 0xFFFFFFFFu);              // global_addr[31:0]
  g0.w = (unsigned)((ga >> 32) & 0x01FFFFFFu) | (2u << 30);  // addr[56:32]|type=2
  i32x8 g1;
  // data_size=2B (code 1) @ [17:16]; pad_enable @20; pad_interval=3 (64B) @[24:22];
  // pad_amount=3 (4 DWORDs = 16B) @[31:25]
  g1[0] = (1 << 16) | (1 << 20) | (3 << 22) | (3 << 25);
  g1[1] = (int)((srcStride & 0xFFFFu) << 16);       // tensor_dim0 low16 @[63:48]
  g1[2] = (int)(((srcStride >> 16) & 0xFFFFu) |     // tensor_dim0 hi @[79:64]
                ((unsigned)M_TOT << 16));           // tensor_dim1 low16 @[95:80]
  g1[3] = (int)(tileW << 16);                       // tile_dim0 @[127:112]
  g1[4] = (int)tileH;                               // tile_dim1 @[143:128]
  g1[5] = (int)srcStride;                           // tensor_dim0_stride[31:0]
  g1[6] = 0;                                        // stride hi / dim1_stride
  g1[7] = 0;
  i32x4 gz = {0, 0, 0, 0};
#if defined(__clang_major__) && __clang_major__ >= 23
  i32x8 gz8 = {0, 0, 0, 0, 0, 0, 0, 0};
  __builtin_amdgcn_tensor_load_to_lds(g0, g1, gz, gz, gz8, 0);
#else
  __builtin_amdgcn_tensor_load_to_lds(g0, g1, gz, gz, 0);
#endif
#else
  (void)gsrc; (void)ldsOff; (void)tileW; (void)tileH; (void)srcStride;
#endif
}

// ---------------------------------------------------------------- f32 -> f16
__global__ void cvt_f32_f16(const float* __restrict__ x, _Float16* __restrict__ y, int n) {
  int i = blockIdx.x * 256 + threadIdx.x;
  if (i < n) y[i] = (_Float16)x[i];
}

// f32 [K x N] -> f16 transposed [N x K] (weights; 1 Mi elements, one-shot)
__global__ void cvt_f32_f16_t(const float* __restrict__ x, _Float16* __restrict__ y) {
  int i = blockIdx.x * 256 + threadIdx.x;           // i = k * N + n
  int k = i >> 10, n = i & (D_MODEL - 1);
  y[(size_t)n * D_MODEL + k] = (_Float16)x[i];
}

// ---------------------------------------------------------------- GEMM
// C[M_TOT x 1024] = A(f16) * Wt(f16, pre-transposed [N x K]) + bias.
// Block tile 128x64, 8 waves, each wave 32x32 (2x2 WMMA tiles), K-step 32.
// A tile staged in LDS by the Tensor Data Mover (TDM pads rows 64B->80B so
// ds_load_b128 fragment reads are bank-conflict-free); B-fragments are two
// contiguous 16B global loads per lane out of the L2-resident weights.
// mode 0: f16 per-head [bh][s][d]; mode 1: f16 per-head [bh][d][s]; mode 2: f32.
__global__ __launch_bounds__(256) void gemm_wmma(const _Float16* __restrict__ A,
                                                 const _Float16* __restrict__ Wt,
                                                 const float* __restrict__ bias,
                                                 void* __restrict__ out, int mode) {
  __shared__ __align__(16) _Float16 As[128 * 40];   // 128x32 (+8 halfs pad/row)
  const int K = D_MODEL, N = D_MODEL;
  int tid  = threadIdx.x;
  int lane = tid & 31, wave = tid >> 5;
  int g = lane >> 4, nl = lane & 15;
  int wm = wave & 3, wn = wave >> 2;                // 4x2 wave grid
  int m0 = blockIdx.x * 128, n0 = blockIdx.y * 64;
  unsigned asOff = (unsigned)(uintptr_t)(&As[0]);   // LDS byte offset

  v8f acc[2][2] = {};

  for (int k0 = 0; k0 < K; k0 += 32) {
    __syncthreads();                                // readers done with LDS tile
#if __has_builtin(__builtin_amdgcn_tensor_load_to_lds)
    if (tid < 32) {                                 // wave 0 drives the TDM
      tdm_load_tile_f16(A + (size_t)m0 * K + k0, asOff, 32, 128, K);
      wait_tensorcnt0();
    }
#else
    {                                               // manual staging fallback
      int row = tid >> 1, ch = (tid & 1) * 16;
      const u32x4* src = (const u32x4*)(A + (size_t)(m0 + row) * K + k0 + ch);
      u32x4* dst = (u32x4*)(&As[row * 40 + ch]);
      dst[0] = src[0];
      dst[1] = src[1];
    }
#endif
    __syncthreads();                                // tile visible to all waves

    v16h af[2], bf[2];
#pragma unroll
    for (int i = 0; i < 2; ++i)
      af[i] = load_a_frag(&As[(wm * 32 + i * 16) * 40], 40, lane);
#pragma unroll
    for (int j = 0; j < 2; ++j)
      bf[j] = load_b_frag(Wt + (size_t)(n0 + wn * 32 + j * 16 + nl) * K + k0 + g * 16);
#pragma unroll
    for (int i = 0; i < 2; ++i)
#pragma unroll
      for (int j = 0; j < 2; ++j)
        acc[i][j] = wmma_f16(af[i], bf[j], acc[i][j]);
  }

  // Epilogue: C layout row = 8*g + vgpr, col = lane%16
#pragma unroll
  for (int i = 0; i < 2; ++i) {
#pragma unroll
    for (int j = 0; j < 2; ++j) {
      int col = n0 + wn * 32 + j * 16 + nl;
      float bv = bias[col];
#pragma unroll
      for (int e = 0; e < 8; ++e) {
        int row = m0 + wm * 32 + i * 16 + g * 8 + e;
        float v = acc[i][j][e] + bv;
        if (mode == 2) {
          ((float*)out)[(size_t)row * N + col] = v;
        } else {
          int bb = row >> 11;          // / SEQ
          int s  = row & (SEQ - 1);
          int hh = col >> 6;           // / DK
          int d  = col & (DK - 1);
          _Float16 hv = (_Float16)v;
          if (mode == 0)
            ((_Float16*)out)[((size_t)(bb * NUM_HEADS + hh) * SEQ + s) * DK + d] = hv;
          else
            ((_Float16*)out)[((size_t)(bb * NUM_HEADS + hh) * DK + d) * SEQ + s] = hv;
        }
      }
    }
  }
}

// ---------------------------------------------------------------- Attention
// Grid (SEQ/128, BATCH*NUM_HEADS), 256 threads. Each wave owns a 16-row Q tile.
// Two-pass softmax: pass1 row max+denominator (online), pass2 recompute scores,
// emit exact normalized probs to attn (f32) once, and accumulate ctx = P @ V.
__global__ __launch_bounds__(256) void attn_wmma(const _Float16* __restrict__ Qp,
                                                 const _Float16* __restrict__ Kp,
                                                 const _Float16* __restrict__ Vpt,
                                                 float* __restrict__ attn,
                                                 _Float16* __restrict__ ctx16) {
  __shared__ __align__(16) _Float16 pLds[8 * 16 * 40];  // per-wave 16x32 (+8 pad)
  const float scale = 0.125f;                           // 1/sqrt(64)
  int lane = threadIdx.x & 31, wave = threadIdx.x >> 5;
  int g = lane >> 4, n = lane & 15;
  int bh = blockIdx.y, b = bh >> 4, h = bh & (NUM_HEADS - 1);
  int qbase = blockIdx.x * 128 + wave * 16;
  int row0 = qbase + g * 8;

  // Q fragments stay in registers for both passes
  const _Float16* qb = Qp + ((size_t)bh * SEQ + qbase) * DK;
  v16h a0 = load_a_frag(qb, DK, lane);        // d = 0..31
  v16h a1 = load_a_frag(qb + 32, DK, lane);   // d = 32..63

  float m[8], l[8];
#pragma unroll
  for (int i = 0; i < 8; ++i) { m[i] = -1e30f; l[i] = 0.f; }

  // ---- pass 1: row max + denominator
  for (int kt = 0; kt < SEQ / 16; ++kt) {
    const _Float16* kp = Kp + ((size_t)bh * SEQ + kt * 16 + n) * DK + g * 16;
    v8f c = {};
    c = wmma_f16(a0, load_b_frag(kp), c);
    c = wmma_f16(a1, load_b_frag(kp + 32), c);
#pragma unroll
    for (int i = 0; i < 8; ++i) {
      float s = c[i] * scale;
      float t = s;
      t = fmaxf(t, __shfl_xor(t, 1)); t = fmaxf(t, __shfl_xor(t, 2));
      t = fmaxf(t, __shfl_xor(t, 4)); t = fmaxf(t, __shfl_xor(t, 8));
      float nm = fmaxf(m[i], t);
      float p  = __expf(s - nm);
      float ps = p;
      ps += __shfl_xor(ps, 1); ps += __shfl_xor(ps, 2);
      ps += __shfl_xor(ps, 4); ps += __shfl_xor(ps, 8);
      l[i] = l[i] * __expf(m[i] - nm) + ps;
      m[i] = nm;
    }
  }
  float rl[8];
#pragma unroll
  for (int i = 0; i < 8; ++i) rl[i] = 1.f / l[i];

  // ---- pass 2: normalized probs out + ctx = P @ V
  v8f acc[4] = {};
  _Float16* pw = pLds + wave * (16 * 40);
  for (int kb = 0; kb < SEQ / 32; ++kb) {
#pragma unroll
    for (int t2 = 0; t2 < 2; ++t2) {
      int kt = kb * 2 + t2;
      const _Float16* kp = Kp + ((size_t)bh * SEQ + kt * 16 + n) * DK + g * 16;
      v8f c = {};
      c = wmma_f16(a0, load_b_frag(kp), c);
      c = wmma_f16(a1, load_b_frag(kp + 32), c);
      int col = kb * 32 + t2 * 16 + n;
#pragma unroll
      for (int i = 0; i < 8; ++i) {
        float p = __expf(c[i] * scale - m[i]) * rl[i];
        attn[((size_t)bh * SEQ + row0 + i) * SEQ + col] = p;     // exact probs
        pw[(g * 8 + i) * 40 + t2 * 16 + n] = (_Float16)p;        // C->A relayout
      }
    }
    // DS ops are issued/processed in order per wave; wait so all lanes' stores
    // are visible before the cross-lane A-fragment reads.
    asm volatile("s_wait_dscnt 0" ::: "memory");
    v16h aP = load_a_frag(pw, 40, lane);                         // P 16x32 f16
#pragma unroll
    for (int j = 0; j < 4; ++j) {
      const _Float16* vb = Vpt + ((size_t)bh * DK + j * 16 + n) * SEQ + kb * 32 + g * 16;
      acc[j] = wmma_f16(aP, load_b_frag(vb), acc[j]);
    }
    asm volatile("s_wait_dscnt 0" ::: "memory");                 // WAR guard
  }

  // ctx (f16, row-major [4096 x 1024]) feeds the output projection
#pragma unroll
  for (int j = 0; j < 4; ++j)
#pragma unroll
    for (int i = 0; i < 8; ++i)
      ctx16[((size_t)b * SEQ + row0 + i) * D_MODEL + h * DK + j * 16 + n] =
          (_Float16)acc[j][i];
}

// ---------------------------------------------------------------- launch
extern "C" void kernel_launch(void* const* d_in, const int* in_sizes, int n_in,
                              void* d_out, int out_size, void* d_ws, size_t ws_size,
                              hipStream_t stream) {
  (void)in_sizes; (void)n_in; (void)out_size; (void)ws_size;
  const float* q  = (const float*)d_in[0];
  const float* k  = (const float*)d_in[1];
  const float* v  = (const float*)d_in[2];
  const float* Wq = (const float*)d_in[3];
  const float* bq = (const float*)d_in[4];
  const float* Wk = (const float*)d_in[5];
  const float* bk = (const float*)d_in[6];
  const float* Wv = (const float*)d_in[7];
  const float* bv = (const float*)d_in[8];
  const float* Wo = (const float*)d_in[9];
  const float* bo = (const float*)d_in[10];

  const size_t ACT = (size_t)M_TOT * D_MODEL;    // 4 Mi elements
  const size_t WEL = (size_t)D_MODEL * D_MODEL;  // 1 Mi elements

  // Workspace layout (f16): 64 MB total
  _Float16* w0   = (_Float16*)d_ws;
  _Float16* q16  = w0;
  _Float16* k16  = q16 + ACT;
  _Float16* v16  = k16 + ACT;
  _Float16* wq16 = v16 + ACT;    // transposed [N x K]
  _Float16* wk16 = wq16 + WEL;   // transposed
  _Float16* wv16 = wk16 + WEL;   // transposed
  _Float16* wo16 = wv16 + WEL;   // transposed
  _Float16* Qp   = wo16 + WEL;   // [bh][s][d]
  _Float16* Kp   = Qp + ACT;     // [bh][s][d]
  _Float16* Vpt  = Kp + ACT;     // [bh][d][s] (transposed for contiguous B-frags)
  _Float16* ctx  = Vpt + ACT;    // [m][n] row-major

  float* outp = (float*)d_out;          // [2,2048,1024]
  float* attn = outp + ACT;             // [2,16,2048,2048]

  // f32 -> f16 conversions (weights transposed for direct B-fragment loads)
  cvt_f32_f16<<<(int)((ACT + 255) / 256), 256, 0, stream>>>(q, q16, (int)ACT);
  cvt_f32_f16<<<(int)((ACT + 255) / 256), 256, 0, stream>>>(k, k16, (int)ACT);
  cvt_f32_f16<<<(int)((ACT + 255) / 256), 256, 0, stream>>>(v, v16, (int)ACT);
  cvt_f32_f16_t<<<(int)((WEL + 255) / 256), 256, 0, stream>>>(Wq, wq16);
  cvt_f32_f16_t<<<(int)((WEL + 255) / 256), 256, 0, stream>>>(Wk, wk16);
  cvt_f32_f16_t<<<(int)((WEL + 255) / 256), 256, 0, stream>>>(Wv, wv16);
  cvt_f32_f16_t<<<(int)((WEL + 255) / 256), 256, 0, stream>>>(Wo, wo16);

  dim3 ggrid(M_TOT / 128, D_MODEL / 64);
  gemm_wmma<<<ggrid, 256, 0, stream>>>(q16, wq16, bq, (void*)Qp, 0);
  gemm_wmma<<<ggrid, 256, 0, stream>>>(k16, wk16, bk, (void*)Kp, 0);
  gemm_wmma<<<ggrid, 256, 0, stream>>>(v16, wv16, bv, (void*)Vpt, 1);

  attn_wmma<<<dim3(SEQ / 128, BATCH * NUM_HEADS), 256, 0, stream>>>(Qp, Kp, Vpt,
                                                                    attn, ctx);

  gemm_wmma<<<ggrid, 256, 0, stream>>>(ctx, wo16, bo, (void*)outp, 2);
}
